// LinearAttention_13022340842148
// MI455X (gfx1250) — compile-verified, matrix operand order checked by reading, not run on hardware
//
#include <hip/hip_runtime.h>

typedef __attribute__((ext_vector_type(16))) _Float16 v16h;
typedef __attribute__((ext_vector_type(8)))  _Float16 v8h;
typedef __attribute__((ext_vector_type(4)))  _Float16 v4h;
typedef __attribute__((ext_vector_type(2)))  _Float16 v2h;
typedef __attribute__((ext_vector_type(8)))  float    v8f;
typedef __attribute__((ext_vector_type(4)))  float    v4f;
typedef __attribute__((ext_vector_type(4)))  unsigned int v4u;
typedef __attribute__((ext_vector_type(8)))  int      v8i_;
typedef __attribute__((ext_vector_type(4)))  int      v4i_;

#define B_SZ    8
#define T_LEN   4096
#define N_EMBD  1024
#define N_HEAD  16
#define HEAD_D  64
#define C3      (3 * N_EMBD)
#define M_ROWS  (B_SZ * T_LEN)     // 32768
#define NCHUNK  16
#define CHUNK   (T_LEN / NCHUNK)   // 256
#define EPSV    1e-3f

// ---------------- GEMM tiling ----------------
#define BM 128
#define BN 128
#define BK 32
#define PAD 8                       // 16B pad per 64B row in LDS
#define GEMM_THREADS 128            // 4 waves, each owns a 64x64 patch (4x4 WMMA tiles)

// ---------------------------------------------------------------------------
// Tensor Data Mover: DMA one 128x32 f16 tile (row stride K elems) into LDS,
// writing the padded row image (64B data + 16B pad) directly via D# padding.
// All descriptor fields are wave-uniform -> SGPRs. Tracked by TENSORcnt.
// ---------------------------------------------------------------------------
__device__ __forceinline__ void tdm_load_tile(const _Float16* gsrc, void* lds_dst,
                                              int tensor_rows, int K) {
    unsigned long long ga = (unsigned long long)gsrc;
    unsigned lds = (unsigned)(unsigned long long)lds_dst;   // low 32 bits = LDS offset

    v4u g0;
    g0.x = 1u;                                              // count=1 (valid D#)
    g0.y = lds;                                             // lds_addr [63:32]
    g0.z = (unsigned)(ga & 0xffffffffu);                    // global_addr [95:64]
    g0.w = (unsigned)((ga >> 32) & 0x01ffffffu)             // global_addr [120:96]
         | (2u << 30);                                      // type=2 ("image")

    v8i_ g1;
    g1[0] = (int)((1u << 16)        // data_size = 2 bytes
                | (1u << 20)        // pad_enable
                | (3u << 22)        // pad_interval: 16 DWORDs (=64B) before pad
                | (3u << 25));      // pad_amount: 4 DWORDs (=16B)
    g1[1] = (int)(((unsigned)K & 0xffffu) << 16);                 // tensor_dim0 lo
    g1[2] = (int)(((unsigned)K >> 16)
                | (((unsigned)tensor_rows & 0xffffu) << 16));     // dim0 hi | dim1 lo
    g1[3] = (int)(((unsigned)tensor_rows >> 16) | (32u << 16));   // dim1 hi | tile_dim0=32
    g1[4] = (int)128u;                                            // tile_dim1=128, tile_dim2=0
    g1[5] = (int)K;                                               // tensor_dim0_stride lo
    g1[6] = 0;                                                    // stride hi | dim1_stride lo
    g1[7] = 0;
    v4i_ gz4 = {0, 0, 0, 0};
    v8i_ gz8 = {0, 0, 0, 0, 0, 0, 0, 0};
    __builtin_amdgcn_tensor_load_to_lds(g0, g1, gz4, gz4, gz8, 0);
}

// ---------------------------------------------------------------------------
// C[m,n] = sum_k A[m,k]*W[n,k] + bias[n]   (A,W f16 row-major; C f32)
// block = 128 threads (4 waves); wave grid 2(M) x 2(N); each wave: 4x4 WMMA tiles
// ---------------------------------------------------------------------------
__global__ __launch_bounds__(GEMM_THREADS)
void wmma_gemm_f16(const _Float16* __restrict__ A, const _Float16* __restrict__ W,
                   const float* __restrict__ bias, float* __restrict__ Cout,
                   int M, int N, int K) {
    __shared__ _Float16 shA[2][BM][BK + PAD];
    __shared__ _Float16 shB[2][BN][BK + PAD];

    const int tid    = threadIdx.x;
    const int lane   = tid & 31;
    const int wid    = tid >> 5;
    const int wave_m = wid >> 1;   // 0..1 -> 64-row patch
    const int wave_n = wid & 1;    // 0..1 -> 64-col patch
    const int lrow   = lane & 15;
    const int kh     = lane >> 4;  // K-half select per ISA 16-bit A/B layout
    const int m0     = blockIdx.y * BM;
    const int n0     = blockIdx.x * BN;

    v8f acc[4][4] = {};
    const int NK = K / BK;

    if (wid == 0) {                                  // wave 0 drives the TDM
        tdm_load_tile(A + (long long)m0 * K, &shA[0][0][0], M, K);
        tdm_load_tile(W + (long long)n0 * K, &shB[0][0][0], N, K);
    }

    for (int it = 0; it < NK; ++it) {
        __builtin_amdgcn_s_wait_tensorcnt((short)0); // non-issuers: count already 0
        __syncthreads();                             // tile resident for all waves
        const int cur = it & 1;
        if (wid == 0 && it + 1 < NK) {               // DMA next tile into other buffer
            const int k1 = (it + 1) * BK;
            tdm_load_tile(A + (long long)m0 * K + k1, &shA[cur ^ 1][0][0], M, K);
            tdm_load_tile(W + (long long)n0 * K + k1, &shB[cur ^ 1][0][0], N, K);
        }

        v16h af[4], bf[4];
#pragma unroll
        for (int tm = 0; tm < 4; ++tm) {
            const _Float16* p = &shA[cur][wave_m * 64 + tm * 16 + lrow][0];
            v8h lo = *(const v8h*)(p + kh * 8);        // K = kh*8 .. kh*8+7
            v8h hi = *(const v8h*)(p + 16 + kh * 8);   // K = 16+kh*8 ..
            af[tm] = __builtin_shufflevector(lo, hi, 0,1,2,3,4,5,6,7,
                                                     8,9,10,11,12,13,14,15);
        }
#pragma unroll
        for (int tn = 0; tn < 4; ++tn) {
            const _Float16* p = &shB[cur][wave_n * 64 + tn * 16 + lrow][0];
            v8h lo = *(const v8h*)(p + kh * 8);
            v8h hi = *(const v8h*)(p + 16 + kh * 8);
            bf[tn] = __builtin_shufflevector(lo, hi, 0,1,2,3,4,5,6,7,
                                                     8,9,10,11,12,13,14,15);
        }
#pragma unroll
        for (int tm = 0; tm < 4; ++tm)
#pragma unroll
            for (int tn = 0; tn < 4; ++tn)
                acc[tm][tn] = __builtin_amdgcn_wmma_f32_16x16x32_f16(
                    false, af[tm], false, bf[tn], (short)0, acc[tm][tn],
                    false, false);
    }

    // epilogue: C layout — lane<16: N=lane, M=r ; lane>=16: N=lane-16, M=8+r
    const int m_hi = kh * 8;
#pragma unroll
    for (int tm = 0; tm < 4; ++tm) {
        const int mb = m0 + wave_m * 64 + tm * 16 + m_hi;
#pragma unroll
        for (int tn = 0; tn < 4; ++tn) {
            const int n  = n0 + wave_n * 64 + tn * 16 + lrow;
            const float bv = bias[n];
            float* op = Cout + (long long)mb * N + n;
#pragma unroll
            for (int r = 0; r < 8; ++r)
                op[(long long)r * N] = acc[tm][tn][r] + bv;
        }
    }
}

// ---------------------------------------------------------------------------
// Linear-attention chunked scan (wave32, lane owns d pair {2*lane, 2*lane+1})
// ---------------------------------------------------------------------------
__device__ __forceinline__ float featmap(float v) {      // elu(v)+1
    return v > 0.f ? v + 1.f : __expf(v);
}

__global__ __launch_bounds__(32)
void la_partial(const float* __restrict__ qkv, const float* __restrict__ scale_p,
                float* __restrict__ ksum, float* __restrict__ kvsum) {
    const int blk = blockIdx.x;
    const int c = blk & (NCHUNK - 1);
    const int h = (blk >> 4) & (N_HEAD - 1);
    const int b = blk >> 8;
    const int d0 = threadIdx.x * 2;
    const float s = *scale_p;
    long long base = ((long long)(b * T_LEN + c * CHUNK)) * C3 + h * HEAD_D + d0;
    float sk0 = 0.f, sk1 = 0.f, sv0 = 0.f, sv1 = 0.f;
    for (int t = 0; t < CHUNK; ++t) {
        float2 kk = *(const float2*)(qkv + base + N_EMBD);
        float2 vv = *(const float2*)(qkv + base + 2 * N_EMBD);
        float k0 = featmap(kk.x * s), k1 = featmap(kk.y * s);
        sk0 += k0; sk1 += k1; sv0 += k0 * vv.x; sv1 += k1 * vv.y;
        base += C3;
    }
    long long o = (((long long)(b * N_HEAD + h)) * NCHUNK + c) * HEAD_D + d0;
    *(float2*)(ksum + o)  = make_float2(sk0, sk1);
    *(float2*)(kvsum + o) = make_float2(sv0, sv1);
}

__global__ __launch_bounds__(32)
void la_prefix(float* __restrict__ ksum, float* __restrict__ kvsum) {
    const int bh = blockIdx.x;
    const int d0 = threadIdx.x * 2;
    float rk0 = 0.f, rk1 = 0.f, rv0 = 0.f, rv1 = 0.f;
    for (int c = 0; c < NCHUNK; ++c) {
        long long o = (((long long)bh) * NCHUNK + c) * HEAD_D + d0;
        float2 a = *(float2*)(ksum + o);
        float2 v = *(float2*)(kvsum + o);
        *(float2*)(ksum + o)  = make_float2(rk0, rk1);   // exclusive prefix
        *(float2*)(kvsum + o) = make_float2(rv0, rv1);
        rk0 += a.x; rk1 += a.y; rv0 += v.x; rv1 += v.y;
    }
}

__global__ __launch_bounds__(32)
void la_scan(const float* __restrict__ qkv, const float* __restrict__ scale_p,
             const float* __restrict__ ksum, const float* __restrict__ kvsum,
             _Float16* __restrict__ yh) {
    const int blk = blockIdx.x;
    const int c = blk & (NCHUNK - 1);
    const int h = (blk >> 4) & (N_HEAD - 1);
    const int b = blk >> 8;
    const int d0 = threadIdx.x * 2;
    const float s = *scale_p;

    long long o = (((long long)(b * N_HEAD + h)) * NCHUNK + c) * HEAD_D + d0;
    float2 isk = *(const float2*)(ksum + o);
    float2 isv = *(const float2*)(kvsum + o);
    float sk0 = isk.x, sk1 = isk.y, sv0 = isv.x, sv1 = isv.y;

    long long base  = ((long long)(b * T_LEN + c * CHUNK)) * C3 + h * HEAD_D + d0;
    long long ybase = ((long long)(b * T_LEN + c * CHUNK)) * N_EMBD + h * HEAD_D + d0;
    for (int t = 0; t < CHUNK; ++t) {
        float2 qq = *(const float2*)(qkv + base);
        float2 kk = *(const float2*)(qkv + base + N_EMBD);
        float2 vv = *(const float2*)(qkv + base + 2 * N_EMBD);
        float k0 = featmap(kk.x * s), k1 = featmap(kk.y * s);
        sk0 += k0; sk1 += k1; sv0 += k0 * vv.x; sv1 += k1 * vv.y;   // inclusive cumsum
        float q0 = featmap(qq.x * s), q1 = featmap(qq.y * s);
        float part = q0 * sk0 + q1 * sk1;
#pragma unroll
        for (int m = 16; m > 0; m >>= 1) part += __shfl_xor(part, m, 32);
        float denom = fmaxf(part, EPSV);
        v2h oy;
        oy.x = (_Float16)(q0 * sv0 / denom);
        oy.y = (_Float16)(q1 * sv1 / denom);
        *(v2h*)(yh + ybase) = oy;
        base += C3; ybase += N_EMBD;
    }
}

// ---------------------------------------------------------------------------
__global__ void cvt_f32_to_f16(const float* __restrict__ in,
                               _Float16* __restrict__ out, long long n4) {
    long long i = (long long)blockIdx.x * blockDim.x + threadIdx.x;
    const long long stride = (long long)gridDim.x * blockDim.x;
    for (; i < n4; i += stride) {
        v4f f = ((const v4f*)in)[i];
        v4h hh;
        hh.x = (_Float16)f.x; hh.y = (_Float16)f.y;
        hh.z = (_Float16)f.z; hh.w = (_Float16)f.w;
        ((v4h*)out)[i] = hh;
    }
}

// ---------------------------------------------------------------------------
extern "C" void kernel_launch(void* const* d_in, const int* in_sizes, int n_in,
                              void* d_out, int out_size, void* d_ws, size_t ws_size,
                              hipStream_t stream) {
    (void)in_sizes; (void)n_in; (void)out_size; (void)ws_size;
    const float* x     = (const float*)d_in[0];
    const float* Wqkv  = (const float*)d_in[1];
    const float* bqkv  = (const float*)d_in[2];
    const float* Wproj = (const float*)d_in[3];
    const float* bproj = (const float*)d_in[4];
    const float* scale = (const float*)d_in[5];
    float* out = (float*)d_out;

    char* ws = (char*)d_ws;
    size_t off = 0;
    auto carve = [&](size_t bytes) -> char* {
        char* p = ws + off;
        off = (off + bytes + 255) & ~(size_t)255;
        return p;
    };
    _Float16* xh    = (_Float16*)carve((size_t)M_ROWS * N_EMBD * 2);   // 64 MB
    _Float16* wqh   = (_Float16*)carve((size_t)C3 * N_EMBD * 2);       // 6 MB
    _Float16* wph   = (_Float16*)carve((size_t)N_EMBD * N_EMBD * 2);   // 2 MB
    float*    qkv   = (float*)carve((size_t)M_ROWS * C3 * 4);          // 384 MB
    _Float16* yh    = (_Float16*)carve((size_t)M_ROWS * N_EMBD * 2);   // 64 MB
    float*    ksum  = (float*)carve((size_t)B_SZ * N_HEAD * NCHUNK * HEAD_D * 4);
    float*    kvsum = (float*)carve((size_t)B_SZ * N_HEAD * NCHUNK * HEAD_D * 4);

    // 1) convert operands to f16
    cvt_f32_to_f16<<<4096, 256, 0, stream>>>(x, xh, (long long)M_ROWS * N_EMBD / 4);
    cvt_f32_to_f16<<<512, 256, 0, stream>>>(Wqkv, wqh, (long long)C3 * N_EMBD / 4);
    cvt_f32_to_f16<<<256, 256, 0, stream>>>(Wproj, wph, (long long)N_EMBD * N_EMBD / 4);

    // 2) qkv = x @ Wqkv^T + bqkv     [32768 x 3072]
    dim3 g1(C3 / BN, M_ROWS / BM);
    wmma_gemm_f16<<<g1, GEMM_THREADS, 0, stream>>>(xh, wqh, bqkv, qkv,
                                                   M_ROWS, C3, N_EMBD);

    // 3) chunked causal linear-attention scan -> yh (f16)
    la_partial<<<B_SZ * N_HEAD * NCHUNK, 32, 0, stream>>>(qkv, scale, ksum, kvsum);
    la_prefix<<<B_SZ * N_HEAD, 32, 0, stream>>>(ksum, kvsum);
    la_scan<<<B_SZ * N_HEAD * NCHUNK, 32, 0, stream>>>(qkv, scale, ksum, kvsum, yh);

    // 4) out = y @ Wproj^T + bproj   [32768 x 1024]
    dim3 g2(N_EMBD / BN, M_ROWS / BM);
    wmma_gemm_f16<<<g2, GEMM_THREADS, 0, stream>>>(yh, wph, bproj, out,
                                                   M_ROWS, N_EMBD, N_EMBD);
}
